// Decoder_39633958207801
// MI455X (gfx1250) — matrix operand order repaired and checked by reading, not compiled, not run on hardware
//
#include <hip/hip_runtime.h>
#include <hip/hip_bf16.h>

typedef __attribute__((ext_vector_type(16))) __bf16 v16bf;
typedef __attribute__((ext_vector_type(2)))  __bf16 v2bf;
typedef __attribute__((ext_vector_type(8)))  float  v8f;

__device__ __forceinline__ __bf16 to_bf16(float f) {
    __hip_bfloat16 h = __float2bfloat16(f);
    return *reinterpret_cast<__bf16*>(&h);
}
__device__ __forceinline__ float sigf(float x) { return 1.0f / (1.0f + __expf(-x)); }

// ---------------------------------------------------------------------------
// bf16 WMMA GEMM, register-blocked:  C[M,N] = act( A[M,K] @ W[N,K]^T + addend + bias )
// A row-major (lda), W row-major (ldb) -> computes A@W.T with f32 accumulate.
// Each wave owns a 32x64 macro-tile: 2 M-tiles x 4 N-tiles -> 8 WMMAs / K-step
// against 6 fragment loads (2 A + 4 B), ~21 FLOP per L2 byte.
// 4 waves / block along N. Requires M % 32 == 0, N % 256 == 0, K % 32 == 0.
// ---------------------------------------------------------------------------
union FragBF { v16bf v; v2bf p[8]; };

__global__ __launch_bounds__(128)
void k_wmma_gemm(const __bf16* __restrict__ A, int lda,
                 const __bf16* __restrict__ W, int ldb,
                 float* __restrict__ Cf, __bf16* __restrict__ Cb, int ldc,
                 const float* __restrict__ addend, int ldadd,
                 const float* __restrict__ bias,
                 int N, int K, int act) {
    const int wave  = threadIdx.x >> 5;
    const int lane  = threadIdx.x & 31;
    const int half  = lane >> 4;          // 0: lanes 0-15, 1: lanes 16-31
    const int r     = lane & 15;
    const int nBase = (blockIdx.x * 4 + wave) * 64;   // 4 N-tiles per wave
    const int mBase = blockIdx.y * 32;                // 2 M-tiles per wave
    if (nBase >= N) return;

    // C/D layout per tile: VGPR i -> row (tileM + half*8 + i), col (tileN + r)
    v8f acc[2][4];
#pragma unroll
    for (int m = 0; m < 2; ++m)
#pragma unroll
        for (int n = 0; n < 4; ++n) {
            if (addend) {
#pragma unroll
                for (int i = 0; i < 8; ++i)
                    acc[m][n][i] = addend[(size_t)(mBase + m * 16 + half * 8 + i) * ldadd
                                          + nBase + n * 16 + r];
            } else {
                acc[m][n] = (v8f){};
            }
        }

    const __bf16* Arow0 = A + (size_t)(mBase + r) * lda;
    const __bf16* Arow1 = A + (size_t)(mBase + 16 + r) * lda;
    const __bf16* Wrow  = W + (size_t)(nBase + r) * ldb;
    const size_t  wstep = (size_t)16 * ldb;

    for (int k = 0; k < K; k += 32) {
        if (k + 32 < K) {
            __builtin_prefetch((const void*)(Wrow + k + 32), 0, 1);
            __builtin_prefetch((const void*)(Arow0 + k + 32), 0, 1);
        }
        FragBF a0, a1;
#pragma unroll
        for (int v = 0; v < 8; ++v) {
            // A 16x32 bf16 layout: lanes<16: K 0-7 (v0-3), 16-23 (v4-7); lanes>=16: +8
            const int ka = ((v & 3) << 1) + ((v >> 2) << 4) + (half << 3);
            a0.p[v] = *(const v2bf*)(Arow0 + k + ka);
            a1.p[v] = *(const v2bf*)(Arow1 + k + ka);
        }
#pragma unroll
        for (int n = 0; n < 4; ++n) {
            FragBF b;
#pragma unroll
            for (int v = 0; v < 8; ++v) {
                // B 32x16 bf16 layout: lanes<16: K 0-7 (v0-3), 16-23 (v4-7); lanes>=16: +8 (col = r)
                const int kb = ((v & 3) << 1) + ((v >> 2) << 4) + (half << 3);
                b.p[v] = *(const v2bf*)(Wrow + (size_t)n * wstep + k + kb);
            }
            acc[0][n] = __builtin_amdgcn_wmma_f32_16x16x32_bf16(
                false, a0.v, false, b.v, (short)0, acc[0][n], false, false);
            acc[1][n] = __builtin_amdgcn_wmma_f32_16x16x32_bf16(
                false, a1.v, false, b.v, (short)0, acc[1][n], false, false);
        }
    }

#pragma unroll
    for (int m = 0; m < 2; ++m)
#pragma unroll
        for (int n = 0; n < 4; ++n) {
            const int col = nBase + n * 16 + r;
            const float bv = bias ? bias[col] : 0.0f;
#pragma unroll
            for (int i = 0; i < 8; ++i) {
                float v = acc[m][n][i] + bv;
                if (act == 1) v = tanhf(v);
                const size_t idx = (size_t)(mBase + m * 16 + half * 8 + i) * ldc + col;
                if (Cf) Cf[idx] = v;
                if (Cb) Cb[idx] = to_bf16(v);
            }
        }
}

// ---------------------------------------------------------------------------
// Helpers
// ---------------------------------------------------------------------------
__global__ void k_cvt(const float* __restrict__ in, __bf16* __restrict__ out, long long n) {
    long long i = (long long)blockIdx.x * blockDim.x + threadIdx.x;
    if (i < n) out[i] = to_bf16(in[i]);
}

__global__ void k_embed(const int* __restrict__ ids, const float* __restrict__ emb,
                        __bf16* __restrict__ xbf, int IN) {
    int i = blockIdx.x * blockDim.x + threadIdx.x;   // bt*IN + k
    int bt = i / IN, k = i - bt * IN;
    xbf[i] = to_bf16(emb[(size_t)ids[bt] * IN + k]);
}

__global__ void k_init(const float* __restrict__ eh, const float* __restrict__ ec,
                       float* __restrict__ h, float* __restrict__ c,
                       __bf16* __restrict__ merged, int H) {
    int i = blockIdx.x * blockDim.x + threadIdx.x;   // b*H + j
    int b = i / H, j = i - b * H;
    float hv = eh[i];
    h[i] = hv; c[i] = ec[i];
    merged[(size_t)b * 3 * H + j] = to_bf16(hv);
}

__global__ void k_bsum(const float* a, const float* b, float* o, int n) {
    int i = blockIdx.x * blockDim.x + threadIdx.x;
    if (i < n) o[i] = a[i] + b[i];
}

__global__ void k_lstm_point(const float* __restrict__ gates,
                             float* __restrict__ h, float* __restrict__ c,
                             __bf16* __restrict__ merged, int H) {
    int i = blockIdx.x * blockDim.x + threadIdx.x;   // b*H + j
    int b = i / H, j = i - b * H;
    const float* g = gates + (size_t)b * 4 * H;
    float ig = sigf(g[j]);
    float fg = sigf(g[j + H]);
    float gg = tanhf(g[j + 2 * H]);
    float og = sigf(g[j + 3 * H]);
    float cn = fg * c[i] + ig * gg;
    float hn = og * tanhf(cn);
    c[i] = cn; h[i] = hn;
    merged[(size_t)b * 3 * H + j] = to_bf16(hn);
}

// aw[b,s] = sum_e q[b,e] * enc[b,s,e]
__global__ void k_attn_scores(const float* __restrict__ q, const float* __restrict__ enc,
                              float* __restrict__ aw, int S, int E2) {
    int s = blockIdx.x, b = blockIdx.y;
    const float* qe = q + (size_t)b * E2;
    const float* ee = enc + ((size_t)b * S + s) * E2;
    float acc = 0.0f;
    for (int e = threadIdx.x; e < E2; e += blockDim.x) acc += qe[e] * ee[e];
    __shared__ float red[256];
    red[threadIdx.x] = acc;
    __syncthreads();
    for (int st = 128; st > 0; st >>= 1) {
        if (threadIdx.x < st) red[threadIdx.x] += red[threadIdx.x + st];
        __syncthreads();
    }
    if (threadIdx.x == 0) aw[b * S + s] = red[0];
}

// merged[b, H+e] = bf16( sum_s aw[b,s] * enc[b,s,e] )
__global__ void k_attn_ctx(const float* __restrict__ aw, const float* __restrict__ enc,
                           __bf16* __restrict__ merged, int S, int E2, int H) {
    int e = blockIdx.x * blockDim.x + threadIdx.x;
    int b = blockIdx.y;
    const float* eb = enc + (size_t)b * S * E2 + e;
    float acc = 0.0f;
    for (int s = 0; s < S; ++s) acc += aw[b * S + s] * eb[(size_t)s * E2];
    merged[(size_t)b * 3 * H + H + e] = to_bf16(acc);
}

__global__ void k_copy(const float* __restrict__ src, float* __restrict__ dst, int n) {
    int i = blockIdx.x * blockDim.x + threadIdx.x;
    if (i < n) dst[i] = src[i];
}

// ---------------------------------------------------------------------------
extern "C" void kernel_launch(void* const* d_in, const int* in_sizes, int n_in,
                              void* d_out, int out_size, void* d_ws, size_t ws_size,
                              hipStream_t stream) {
    const int B = 32, T = 64, S = 128, H = 1024, IN = 512, V = 32000;
    const int E2 = 2 * H, G = 4 * H, M3 = 3 * H;

    const int*   inputs    = (const int*)  d_in[0];
    const float* enc       = (const float*)d_in[1];
    const float* enc_h     = (const float*)d_in[2];
    const float* enc_c     = (const float*)d_in[3];
    const float* emb       = (const float*)d_in[4];
    const float* w_ih      = (const float*)d_in[5];
    const float* w_hh      = (const float*)d_in[6];
    const float* b_ih      = (const float*)d_in[7];
    const float* b_hh      = (const float*)d_in[8];
    const float* attn_W    = (const float*)d_in[9];
    const float* combine_W = (const float*)d_in[10];
    const float* combine_b = (const float*)d_in[11];
    const float* out_W     = (const float*)d_in[12];
    const float* out_b     = (const float*)d_in[13];
    float* out = (float*)d_out;

    char* wsp = (char*)d_ws;
    auto alloc = [&](size_t bytes) -> char* {
        char* p = wsp;
        wsp += (bytes + 255) & ~(size_t)255;
        return p;
    };
    __bf16* w_ih_bf  = (__bf16*)alloc((size_t)G * IN * 2);
    __bf16* w_hh_bf  = (__bf16*)alloc((size_t)G * H * 2);
    __bf16* attn_bf  = (__bf16*)alloc((size_t)E2 * H * 2);
    __bf16* comb_bf  = (__bf16*)alloc((size_t)H * M3 * 2);
    __bf16* outW_bf  = (__bf16*)alloc((size_t)V * H * 2);
    __bf16* x_bf     = (__bf16*)alloc((size_t)B * T * IN * 2);
    float*  xproj    = (float*) alloc((size_t)B * T * G * 4);
    float*  bsum     = (float*) alloc((size_t)G * 4);
    float*  h        = (float*) alloc((size_t)B * H * 4);
    float*  c        = (float*) alloc((size_t)B * H * 4);
    float*  gates    = (float*) alloc((size_t)B * G * 4);
    __bf16* merged   = (__bf16*)alloc((size_t)B * M3 * 2);
    float*  q        = (float*) alloc((size_t)B * E2 * 4);
    float*  aw       = (float*) alloc((size_t)B * S * 4);
    __bf16* outs_bf  = (__bf16*)alloc((size_t)B * T * H * 2);

    auto cvt = [&](const float* src, __bf16* dst, long long n) {
        k_cvt<<<dim3((unsigned)((n + 255) / 256)), 256, 0, stream>>>(src, dst, n);
    };
    auto gemm = [&](const __bf16* A, int lda, const __bf16* Wm, int ldb,
                    float* Cf, __bf16* Cb, int ldc,
                    const float* addend, int ldadd, const float* bias,
                    int M, int N, int K, int act) {
        dim3 grid(N / 256, M / 32);   // wave macro-tile 32x64, 4 waves/block
        k_wmma_gemm<<<grid, 128, 0, stream>>>(A, lda, Wm, ldb, Cf, Cb, ldc,
                                              addend, ldadd, bias, N, K, act);
    };

    // ---- one-time setup -------------------------------------------------
    cvt(w_ih,      w_ih_bf, (long long)G * IN);
    cvt(w_hh,      w_hh_bf, (long long)G * H);
    cvt(attn_W,    attn_bf, (long long)E2 * H);
    cvt(combine_W, comb_bf, (long long)H * M3);
    cvt(out_W,     outW_bf, (long long)V * H);
    k_bsum<<<G / 256, 256, 0, stream>>>(b_ih, b_hh, bsum, G);
    k_embed<<<(B * T * IN) / 256, 256, 0, stream>>>(inputs, emb, x_bf, IN);
    k_init<<<(B * H) / 256, 256, 0, stream>>>(enc_h, enc_c, h, c, merged, H);

    // xproj[B*T, 4H] = emb[inputs] @ w_ih.T + (b_ih + b_hh)   (parallel over T)
    gemm(x_bf, IN, w_ih_bf, IN, xproj, nullptr, G,
         nullptr, 0, bsum, B * T, G, IN, 0);

    // ---- recurrent loop -------------------------------------------------
    for (int t = 0; t < T; ++t) {
        // gates[B,4H] = xproj[:,t,:] + h @ w_hh.T   (addend loaded into WMMA acc)
        gemm(merged, M3, w_hh_bf, H, gates, nullptr, G,
             xproj + (size_t)t * G, T * G, nullptr, B, G, H, 0);
        // LSTM cell pointwise; writes h, c, and h->merged[:, 0:H] (bf16)
        k_lstm_point<<<(B * H) / 256, 256, 0, stream>>>(gates, h, c, merged, H);
        // q[B,2H] = h @ attn_W.T
        gemm(merged, M3, attn_bf, H, q, nullptr, E2,
             nullptr, 0, nullptr, B, E2, H, 0);
        // aw[b,s] = <q[b], enc[b,s]> ; ap -> merged[:, H:3H] (bf16)
        k_attn_scores<<<dim3(S, B), 256, 0, stream>>>(q, enc, aw, S, E2);
        k_attn_ctx<<<dim3(E2 / 256, B), 256, 0, stream>>>(aw, enc, merged, S, E2, H);
        // outs[:,t,:] = tanh(merged @ combine_W.T + b)  -> bf16 rows at stride T*H
        gemm(merged, M3, comb_bf, M3, nullptr, outs_bf + (size_t)t * H, T * H,
             nullptr, 0, combine_b, B, H, M3, 1);
    }

    // ---- final projection: logits[B*T, V] = outs @ out_W.T + out_b ------
    gemm(outs_bf, H, outW_bf, H, out, nullptr, V,
         nullptr, 0, out_b, B * T, V, H, 0);

    // ---- append h, c to output ------------------------------------------
    const size_t logitsN = (size_t)B * T * V;
    k_copy<<<(B * H) / 256, 256, 0, stream>>>(h, out + logitsN, B * H);
    k_copy<<<(B * H) / 256, 256, 0, stream>>>(c, out + logitsN + B * H, B * H);
}
